// GAAE_18236431138889
// MI455X (gfx1250) — compile-verified, matrix operand order checked by reading, not run on hardware
//
#include <hip/hip_runtime.h>
#include <hip/hip_bf16.h>

// GAAE (ChebConv K=2 autoencoder) for MI455X / gfx1250.
// Dense transforms use v_wmma_f32_16x16x32_bf16 (one 16x16 tile per wave32),
// templated on <FIN,FOUT> so K-loops fully unroll and all padding guards fold
// to compile-time constants (no EXEC-mask predication around fragment loads).
// Sparse aggregation uses global_atomic_add_f32; BN uses ds_add_f32 LDS
// accumulation. All activations fit in the 192MB L2, so the scatter phase is
// L2-resident; the GEMM phase (3.7 GFLOP) is pure WMMA.

#define NN 50000
#define EE 1600000
#define FF 128
#define BN_EPS 1e-5f

typedef __attribute__((ext_vector_type(16))) __bf16 v16bf;
typedef __attribute__((ext_vector_type(8)))  float  v8f;
typedef __attribute__((ext_vector_type(8)))  __bf16 bf16x8;

// ---------------------------------------------------------------- reductions
__global__ void gaae_sum_reduce(const float* __restrict__ x, float* __restrict__ out,
                                long long n) {
    __shared__ float sdata[256];
    long long stride = (long long)gridDim.x * blockDim.x;
    long long i = (long long)blockIdx.x * blockDim.x + threadIdx.x;
    float s = 0.0f;
    for (; i < n; i += stride) s += x[i];
    sdata[threadIdx.x] = s;
    __syncthreads();
    for (int off = 128; off > 0; off >>= 1) {
        if ((int)threadIdx.x < off) sdata[threadIdx.x] += sdata[threadIdx.x + off];
        __syncthreads();
    }
    if (threadIdx.x == 0) atomicAdd(out, sdata[0]);
}

__global__ void gaae_inject(const float* __restrict__ x, const float* __restrict__ noise,
                            const float* __restrict__ sumPtr, float* __restrict__ xw,
                            long long n, float invNF) {
    long long i = (long long)blockIdx.x * blockDim.x + threadIdx.x;
    if (i >= n) return;
    float mean = sumPtr[0] * invNF;
    xw[i] = x[i] + noise[i] * mean;
}

// ---------------------------------------------------------------- graph norm
__global__ void gaae_degree(const long long* __restrict__ ei, float* __restrict__ deg,
                            int E) {
    int e = blockIdx.x * blockDim.x + threadIdx.x;
    if (e >= E) return;
    int d = (int)ei[(size_t)E + e];          // dst row
    atomicAdd(&deg[d], 1.0f);
}

__global__ void gaae_dinv(float* __restrict__ deg, int n) {
    int i = blockIdx.x * blockDim.x + threadIdx.x;
    if (i >= n) return;
    float d = deg[i];
    deg[i] = (d > 0.0f) ? rsqrtf(d) : 0.0f;  // deg >= 1 when > 0
}

// tx[dst] += (-dinv[src]*dinv[dst]) * h[src]   (one thread per edge x float4)
__global__ void gaae_spmm(const long long* __restrict__ ei, const float* __restrict__ dinv,
                          const float* __restrict__ h, float* __restrict__ tx,
                          int E, int qshift) {
    int quads = 1 << qshift;                 // Fin/4
    long long total = (long long)E << qshift;
    long long idx = (long long)blockIdx.x * blockDim.x + threadIdx.x;
    if (idx >= total) return;
    int e = (int)(idx >> qshift);
    int q = (int)(idx & (quads - 1));
    int s = (int)ei[e];
    int d = (int)ei[(size_t)E + e];
    float w = -dinv[s] * dinv[d];
    size_t fin = (size_t)quads << 2;
    const float4 hv = *(const float4*)(h + (size_t)s * fin + (size_t)q * 4);
    float* o = tx + (size_t)d * fin + (size_t)q * 4;
    atomicAdd(o + 0, w * hv.x);
    atomicAdd(o + 1, w * hv.y);
    atomicAdd(o + 2, w * hv.z);
    atomicAdd(o + 3, w * hv.w);
}

// ------------------------------------------------------- weight prep (WT bf16)
// WT[n*Fin + k] = (bf16) W[k*Fout + n]   (transpose so B-fragment loads are rows)
__global__ void gaae_wt_convert(const float* __restrict__ W, __bf16* __restrict__ WT,
                                int Fin, int Fout) {
    int i = blockIdx.x * blockDim.x + threadIdx.x;
    if (i >= Fin * Fout) return;
    int n = i / Fin;
    int k = i - n * Fin;
    WT[i] = (__bf16)W[k * Fout + n];
}

// ---------------------------------------------------------------- WMMA GEMM
// 16-bit A fragment (16x32): lane = {m = lane&15, half = lane>>4};
// a[0..7]  <- A[m][kk +      half*8 + 0..7]   (valid iff kk    < FIN)
// a[8..15] <- A[m][kk + 16 + half*8 + 0..7]   (valid iff kk+16 < FIN)
// Validity is per 16-aligned K-block, independent of `half` -> wave-uniform
// and compile-time after unrolling (FIN is a template constant).
__device__ __forceinline__ void gaae_loadA_run(const float* p, bool valid,
                                               v16bf& a, int base) {
    if (valid) {
        float4 u = *(const float4*)(p);
        float4 v = *(const float4*)(p + 4);
        a[base + 0] = (__bf16)u.x; a[base + 1] = (__bf16)u.y;
        a[base + 2] = (__bf16)u.z; a[base + 3] = (__bf16)u.w;
        a[base + 4] = (__bf16)v.x; a[base + 5] = (__bf16)v.y;
        a[base + 6] = (__bf16)v.z; a[base + 7] = (__bf16)v.w;
    } else {
#pragma unroll
        for (int i = 0; i < 8; ++i) a[base + i] = (__bf16)0.0f;
    }
}

__device__ __forceinline__ void gaae_loadB_run(const __bf16* p, bool valid,
                                               v16bf& b, int base) {
    if (valid) {
        bf16x8 u = *(const bf16x8*)(p);
#pragma unroll
        for (int i = 0; i < 8; ++i) b[base + i] = u[i];
    } else {
#pragma unroll
        for (int i = 0; i < 8; ++i) b[base + i] = (__bf16)0.0f;
    }
}

// out[N x FOUT] = A@W0 + T@W1 + bias ; one wave per 16x16 tile.
template <int FIN, int FOUT>
__global__ void gaae_gemm2_wmma(const float* __restrict__ A, const float* __restrict__ T,
                                const __bf16* __restrict__ WT0, const __bf16* __restrict__ WT1,
                                const float* __restrict__ bias, float* __restrict__ out,
                                int rowTiles) {
    constexpr int colTiles = FOUT >> 4;
    const int wavesPerBlk = blockDim.x >> 5;
    int wid  = blockIdx.x * wavesPerBlk + ((int)threadIdx.x >> 5);
    int lane = (int)threadIdx.x & 31;
    if (wid >= rowTiles * colTiles) return;   // wave-uniform: EXEC stays all-1s
    int rt   = wid / colTiles;
    int ct   = wid - rt * colTiles;
    int m    = lane & 15;
    int half = lane >> 4;
    int hoff = half * 8;

    const float*  arow  = A + (size_t)(rt * 16 + m) * FIN;
    const float*  trow  = T + (size_t)(rt * 16 + m) * FIN;
    int nCol = ct * 16 + m;
    const __bf16* w0row = WT0 + (size_t)nCol * FIN;
    const __bf16* w1row = WT1 + (size_t)nCol * FIN;

    v8f c = {0.f, 0.f, 0.f, 0.f, 0.f, 0.f, 0.f, 0.f};

#pragma unroll
    for (int kk = 0; kk < FIN; kk += 32) {
        constexpr int dummy = 0; (void)dummy;
        const bool run0 = true;               // kk < FIN inside loop bound
        const bool run1 = (kk + 16) < FIN;    // constant after unroll
        v16bf a = {}, b = {};
        gaae_loadA_run(arow + kk + hoff,      run0, a, 0);
        gaae_loadA_run(arow + kk + 16 + hoff, run1, a, 8);
        gaae_loadB_run(w0row + kk + hoff,      run0, b, 0);
        gaae_loadB_run(w0row + kk + 16 + hoff, run1, b, 8);
        c = __builtin_amdgcn_wmma_f32_16x16x32_bf16(false, a, false, b,
                                                    (short)0, c, false, false);
    }
#pragma unroll
    for (int kk = 0; kk < FIN; kk += 32) {
        const bool run0 = true;
        const bool run1 = (kk + 16) < FIN;
        v16bf a = {}, b = {};
        gaae_loadA_run(trow + kk + hoff,      run0, a, 0);
        gaae_loadA_run(trow + kk + 16 + hoff, run1, a, 8);
        gaae_loadB_run(w1row + kk + hoff,      run0, b, 0);
        gaae_loadB_run(w1row + kk + 16 + hoff, run1, b, 8);
        c = __builtin_amdgcn_wmma_f32_16x16x32_bf16(false, a, false, b,
                                                    (short)0, c, false, false);
    }

    float bv = bias[nCol];
#pragma unroll
    for (int r = 0; r < 8; ++r) {            // D layout: VGPR r -> M = r + 8*half
        int row = rt * 16 + r + half * 8;
        out[(size_t)row * FOUT + nCol] = c[r] + bv;
    }
}

// ---------------------------------------------------------------- batch norm
// stats[c] += sum(h[:,c]); stats[64+c] += sum(h[:,c]^2)  (64-channel BN)
__global__ void gaae_bn_stats(const float* __restrict__ h, float* __restrict__ stats,
                              long long total) {
    __shared__ float ls[128];
    if (threadIdx.x < 128) ls[threadIdx.x] = 0.0f;
    __syncthreads();
    long long stride = (long long)gridDim.x * blockDim.x;
    long long i = (long long)blockIdx.x * blockDim.x + threadIdx.x;
    for (; i < total; i += stride) {
        float v = h[i];
        int c = (int)(i & 63);
        atomicAdd(&ls[c], v);                // ds_add_f32
        atomicAdd(&ls[64 + c], v * v);
    }
    __syncthreads();
    if (threadIdx.x < 128) atomicAdd(&stats[threadIdx.x], ls[threadIdx.x]);
}

__global__ void gaae_bn_apply(float* __restrict__ h, const float* __restrict__ stats,
                              const float* __restrict__ g, const float* __restrict__ be,
                              float invN, long long total) {
    long long i = (long long)blockIdx.x * blockDim.x + threadIdx.x;
    if (i >= total) return;
    int c = (int)(i & 63);
    float mu   = stats[c] * invN;
    float var  = stats[64 + c] * invN - mu * mu;
    float rinv = rsqrtf(var + BN_EPS);
    float v = g[c] * (h[i] - mu) * rinv + be[c];
    h[i] = fmaxf(v, 0.0f);                   // both BN sites are followed by ReLU
}

// ---------------------------------------------------------------- launcher
extern "C" void kernel_launch(void* const* d_in, const int* in_sizes, int n_in,
                              void* d_out, int out_size, void* d_ws, size_t ws_size,
                              hipStream_t stream) {
    const int N = NN, E = EE, F = FF;
    const float*     x     = (const float*)d_in[0];
    const float*     noise = (const float*)d_in[1];
    const long long* ei    = (const long long*)d_in[2];
    const float *W1_0 = (const float*)d_in[3],  *W1_1 = (const float*)d_in[4],
                *b1   = (const float*)d_in[5],  *g1   = (const float*)d_in[6],
                *be1  = (const float*)d_in[7];
    const float *W2_0 = (const float*)d_in[8],  *W2_1 = (const float*)d_in[9],
                *b2   = (const float*)d_in[10];
    const float *W3_0 = (const float*)d_in[11], *W3_1 = (const float*)d_in[12],
                *b3   = (const float*)d_in[13], *g3   = (const float*)d_in[14],
                *be3  = (const float*)d_in[15];
    const float *W4_0 = (const float*)d_in[16], *W4_1 = (const float*)d_in[17],
                *b4   = (const float*)d_in[18];
    float* out = (float*)d_out;

    // ---- workspace carve (~64.3 MB) ----
    char* wsb = (char*)d_ws;
    size_t off = 0;
    auto carve = [&](size_t bytes) -> void* {
        void* p = wsb + off;
        off += (bytes + 255) & ~(size_t)255;
        return p;
    };
    float*  dinv = (float*)carve((size_t)N * 4);          // degree -> dinv in place
    float*  sbuf = (float*)carve(1024);                   // [0]=sum(x); [64..191]=bn stats
    float*  hA   = (float*)carve((size_t)N * 128 * 4);    // xw / h2
    float*  hB   = (float*)carve((size_t)N * 64 * 4);     // h1 / h3
    float*  tx   = (float*)carve((size_t)N * 128 * 4);    // cheb T1 accumulator
    __bf16* wt   = (__bf16*)carve((size_t)36864 * 2);     // transposed bf16 weights
    float*  meanSum = sbuf;
    float*  stats   = sbuf + 64;
    __bf16 *wt1_0 = wt,         *wt1_1 = wt + 8192,
           *wt2_0 = wt + 16384, *wt2_1 = wt + 17408,
           *wt3_0 = wt + 18432, *wt3_1 = wt + 19456,
           *wt4_0 = wt + 20480, *wt4_1 = wt + 28672;

    const int rowTiles = N / 16;             // 3125 exactly
    const long long nf = (long long)N * F;

    // ---- prep: mean(x), degree, dinv, weight convert ----
    hipMemsetAsync(dinv, 0, (size_t)N * 4, stream);
    hipMemsetAsync(meanSum, 0, 4, stream);
    gaae_sum_reduce<<<1024, 256, 0, stream>>>(x, meanSum, nf);
    gaae_degree<<<(E + 255) / 256, 256, 0, stream>>>(ei, dinv, E);
    gaae_dinv<<<(N + 255) / 256, 256, 0, stream>>>(dinv, N);
    gaae_wt_convert<<<(8192 + 255) / 256, 256, 0, stream>>>(W1_0, wt1_0, 128, 64);
    gaae_wt_convert<<<(8192 + 255) / 256, 256, 0, stream>>>(W1_1, wt1_1, 128, 64);
    gaae_wt_convert<<<(1024 + 255) / 256, 256, 0, stream>>>(W2_0, wt2_0, 64, 16);
    gaae_wt_convert<<<(1024 + 255) / 256, 256, 0, stream>>>(W2_1, wt2_1, 64, 16);
    gaae_wt_convert<<<(1024 + 255) / 256, 256, 0, stream>>>(W3_0, wt3_0, 16, 64);
    gaae_wt_convert<<<(1024 + 255) / 256, 256, 0, stream>>>(W3_1, wt3_1, 16, 64);
    gaae_wt_convert<<<(8192 + 255) / 256, 256, 0, stream>>>(W4_0, wt4_0, 64, 128);
    gaae_wt_convert<<<(8192 + 255) / 256, 256, 0, stream>>>(W4_1, wt4_1, 64, 128);
    gaae_inject<<<(int)((nf + 255) / 256), 256, 0, stream>>>(
        x, noise, meanSum, hA, nf, 1.0f / (float)nf);

    // ---- layer 1: 128 -> 64, BN+ReLU ----
    hipMemsetAsync(tx, 0, (size_t)N * 128 * 4, stream);
    gaae_spmm<<<(int)((((long long)E << 5) + 255) / 256), 256, 0, stream>>>(
        ei, dinv, hA, tx, E, 5);
    gaae_gemm2_wmma<128, 64><<<(rowTiles * 4 + 7) / 8, 256, 0, stream>>>(
        hA, tx, wt1_0, wt1_1, b1, hB, rowTiles);
    hipMemsetAsync(stats, 0, 128 * 4, stream);
    gaae_bn_stats<<<1024, 256, 0, stream>>>(hB, stats, (long long)N * 64);
    gaae_bn_apply<<<(int)(((long long)N * 64 + 255) / 256), 256, 0, stream>>>(
        hB, stats, g1, be1, 1.0f / (float)N, (long long)N * 64);

    // ---- layer 2: 64 -> 16 ----
    hipMemsetAsync(tx, 0, (size_t)N * 64 * 4, stream);
    gaae_spmm<<<(int)((((long long)E << 4) + 255) / 256), 256, 0, stream>>>(
        ei, dinv, hB, tx, E, 4);
    gaae_gemm2_wmma<64, 16><<<(rowTiles * 1 + 7) / 8, 256, 0, stream>>>(
        hB, tx, wt2_0, wt2_1, b2, hA, rowTiles);

    // ---- layer 3: 16 -> 64, BN+ReLU ----
    hipMemsetAsync(tx, 0, (size_t)N * 16 * 4, stream);
    gaae_spmm<<<(int)((((long long)E << 2) + 255) / 256), 256, 0, stream>>>(
        ei, dinv, hA, tx, E, 2);
    gaae_gemm2_wmma<16, 64><<<(rowTiles * 4 + 7) / 8, 256, 0, stream>>>(
        hA, tx, wt3_0, wt3_1, b3, hB, rowTiles);
    hipMemsetAsync(stats, 0, 128 * 4, stream);
    gaae_bn_stats<<<1024, 256, 0, stream>>>(hB, stats, (long long)N * 64);
    gaae_bn_apply<<<(int)(((long long)N * 64 + 255) / 256), 256, 0, stream>>>(
        hB, stats, g3, be3, 1.0f / (float)N, (long long)N * 64);

    // ---- layer 4: 64 -> 128 -> d_out ----
    hipMemsetAsync(tx, 0, (size_t)N * 64 * 4, stream);
    gaae_spmm<<<(int)((((long long)E << 4) + 255) / 256), 256, 0, stream>>>(
        ei, dinv, hB, tx, E, 4);
    gaae_gemm2_wmma<64, 128><<<(rowTiles * 8 + 7) / 8, 256, 0, stream>>>(
        hB, tx, wt4_0, wt4_1, b4, out, rowTiles);
}
